// Attention_34840774705284
// MI455X (gfx1250) — compile-verified, matrix operand order checked by reading, not run on hardware
//
#include <hip/hip_runtime.h>

typedef _Float16 h16;
typedef __attribute__((ext_vector_type(16))) _Float16 v16h;
typedef __attribute__((ext_vector_type(8)))  _Float16 v8h;
typedef __attribute__((ext_vector_type(8)))  float    v8f;
typedef __attribute__((ext_vector_type(4)))  int      v4i;

constexpr int Bc = 8;      // batches
constexpr int Nc = 2048;   // kv length
constexpr int Cc = 512;    // channels
constexpr int Hc = 8;      // heads
constexpr int Dc = 64;     // head dim

#if __has_builtin(__builtin_amdgcn_global_load_async_to_lds_b128)
#define HAVE_ASYNC_LDS 1
#else
#define HAVE_ASYNC_LDS 0
#endif

// ---------------------------------------------------------------------------
// WMMA helpers
// ---------------------------------------------------------------------------
__device__ __forceinline__ v8f wmma16(v16h a, v16h b, v8f c) {
  return __builtin_amdgcn_wmma_f32_16x16x32_f16(false, a, false, b, (short)0, c,
                                                false, false);
}

// Per-lane base pointer for a 16-bit fragment (16 rows x 32 K) in a row-major
// matrix with leading dimension ld.  Lane L holds row (L&15); its 16 f16
// K-values are two contiguous 16B chunks at K = 8*(L>>4) and K = 16+8*(L>>4).
// All B operands are stored transposed (NT GEMM) so this serves both A and B.
__device__ __forceinline__ const h16* frag_base(const h16* p, int ld, int row0,
                                                int col0, int lane) {
  return p + (size_t)(row0 + (lane & 15)) * ld + col0 + 8 * (lane >> 4);
}
__device__ __forceinline__ v16h load_frag_at(const h16* base) {
  v8h lo = *(const v8h*)(base);
  v8h hi = *(const v8h*)(base + 16);
  v16h f;
#pragma unroll
  for (int i = 0; i < 8; ++i) { f[i] = lo[i]; f[i + 8] = hi[i]; }
  return f;
}
__device__ __forceinline__ v16h load_frag(const h16* p, int ld, int row0,
                                          int col0, int lane) {
  return load_frag_at(frag_base(p, ld, row0, col0, lane));
}

__device__ __forceinline__ float red16_max(float v) {
#pragma unroll
  for (int m = 8; m >= 1; m >>= 1) v = fmaxf(v, __shfl_xor(v, m, 16));
  return v;
}
__device__ __forceinline__ float red16_sum(float v) {
#pragma unroll
  for (int m = 8; m >= 1; m >>= 1) v += __shfl_xor(v, m, 16);
  return v;
}

// 16B global -> LDS copy: async DMA on gfx1250 (ASYNCcnt-tracked), fallback
// is a synchronous load + ds_store (still shares data via LDS).
// Builtin signature (from compiler diagnostic): (int4 AS1*, int4 AS3*, Ii, Ii)
__device__ __forceinline__ void cp16_to_lds(const h16* g, h16* l) {
#if HAVE_ASYNC_LDS
  __builtin_amdgcn_global_load_async_to_lds_b128(
      (__attribute__((address_space(1))) v4i*)(v4i*)g,
      (__attribute__((address_space(3))) v4i*)(v4i*)l, 0, 0);
#else
  *(v8h*)l = *(const v8h*)g;
#endif
}

template <int N>
__device__ __forceinline__ void wait_asynccnt() {
#if HAVE_ASYNC_LDS
#if __has_builtin(__builtin_amdgcn_s_wait_asynccnt)
  __builtin_amdgcn_s_wait_asynccnt(N);
#else
  asm volatile("s_wait_asynccnt %0" ::"i"(N) : "memory");
#endif
#endif
}

// ---------------------------------------------------------------------------
// f32 -> f16 conversion (grid-stride)
// ---------------------------------------------------------------------------
__global__ void f32_to_f16_kernel(const float* __restrict__ in,
                                  h16* __restrict__ out, int n) {
  int i = blockIdx.x * blockDim.x + threadIdx.x;
  const int stride = gridDim.x * blockDim.x;
  for (; i < n; i += stride) out[i] = (h16)in[i];
}

// ---------------------------------------------------------------------------
// NT GEMM:  C[M x N] = alpha * (A[M x K] * B[N x K]^T)  (+ bias, f32 out)
// Block = 256 threads = 8 waves; wave tile 32x64 (8 accumulators), block tile
// 128x128.  Six per-lane fragment base pointers are fixed; the K offset goes
// into the load's 24-bit immediate, so the pipelined loop has no 64-bit
// address math.  launch_bounds(...,1) lets the allocator keep the 12 in-flight
// fragments in VGPRs instead of spilling to scratch.
// ---------------------------------------------------------------------------
__global__ __launch_bounds__(256, 1) void gemm_nt_wmma(
    const h16* __restrict__ A, int lda, const h16* __restrict__ Bm, int ldb,
    h16* __restrict__ Cf16, float* __restrict__ Cf32, int ldc,
    const float* __restrict__ bias, int M, int N, int K, float alpha) {
  const int lane = threadIdx.x & 31;
  const int wave = threadIdx.x >> 5;
  const int wm = wave & 3, wn = wave >> 2;
  const int row0 = blockIdx.x * 128 + wm * 32;
  const int col0 = blockIdx.y * 128 + wn * 64;
  if (row0 >= M || col0 >= N) return;

  const h16* pA0 = frag_base(A, lda, row0,      0, lane);
  const h16* pA1 = frag_base(A, lda, row0 + 16, 0, lane);
  const h16* pB0 = frag_base(Bm, ldb, col0,      0, lane);
  const h16* pB1 = frag_base(Bm, ldb, col0 + 16, 0, lane);
  const h16* pB2 = frag_base(Bm, ldb, col0 + 32, 0, lane);
  const h16* pB3 = frag_base(Bm, ldb, col0 + 48, 0, lane);

  v8f zero = {};
  v8f acc[2][4] = {{zero, zero, zero, zero}, {zero, zero, zero, zero}};

  v16h a0 = load_frag_at(pA0);
  v16h a1 = load_frag_at(pA1);
  v16h b0 = load_frag_at(pB0);
  v16h b1 = load_frag_at(pB1);
  v16h b2 = load_frag_at(pB2);
  v16h b3 = load_frag_at(pB3);

#pragma unroll 4
  for (int k0 = 32; k0 < K; k0 += 32) {
    v16h na0 = load_frag_at(pA0 + k0);
    v16h na1 = load_frag_at(pA1 + k0);
    v16h nb0 = load_frag_at(pB0 + k0);
    v16h nb1 = load_frag_at(pB1 + k0);
    v16h nb2 = load_frag_at(pB2 + k0);
    v16h nb3 = load_frag_at(pB3 + k0);
    acc[0][0] = wmma16(a0, b0, acc[0][0]);
    acc[0][1] = wmma16(a0, b1, acc[0][1]);
    acc[0][2] = wmma16(a0, b2, acc[0][2]);
    acc[0][3] = wmma16(a0, b3, acc[0][3]);
    acc[1][0] = wmma16(a1, b0, acc[1][0]);
    acc[1][1] = wmma16(a1, b1, acc[1][1]);
    acc[1][2] = wmma16(a1, b2, acc[1][2]);
    acc[1][3] = wmma16(a1, b3, acc[1][3]);
    a0 = na0; a1 = na1; b0 = nb0; b1 = nb1; b2 = nb2; b3 = nb3;
  }
  acc[0][0] = wmma16(a0, b0, acc[0][0]);
  acc[0][1] = wmma16(a0, b1, acc[0][1]);
  acc[0][2] = wmma16(a0, b2, acc[0][2]);
  acc[0][3] = wmma16(a0, b3, acc[0][3]);
  acc[1][0] = wmma16(a1, b0, acc[1][0]);
  acc[1][1] = wmma16(a1, b1, acc[1][1]);
  acc[1][2] = wmma16(a1, b2, acc[1][2]);
  acc[1][3] = wmma16(a1, b3, acc[1][3]);

  const int hl = lane >> 4, cn = lane & 15;
#pragma unroll
  for (int mi = 0; mi < 2; ++mi) {
#pragma unroll
    for (int j = 0; j < 4; ++j) {
#pragma unroll
      for (int r = 0; r < 8; ++r) {
        const int row = row0 + 16 * mi + r + 8 * hl;
        const int col = col0 + 16 * j + cn;
        const float v = acc[mi][j][r] * alpha;
        if (Cf16) {
          Cf16[(size_t)row * ldc + col] = (h16)v;
        } else {
          Cf32[(size_t)row * ldc + col] = v + (bias ? bias[col] : 0.0f);
        }
      }
    }
  }
}

// ---------------------------------------------------------------------------
// KV projection: kv[b] = x[b] @ Wkv^T, scattering K -> (B,H,N,D) row-major
// and V -> (B,H,D,N) transposed so the PV matmul is also NT-form.
// Same structure as gemm_nt_wmma.
// ---------------------------------------------------------------------------
__global__ __launch_bounds__(256, 1) void kv_proj_wmma(
    const h16* __restrict__ x16, const h16* __restrict__ Wkv16,
    h16* __restrict__ k16, h16* __restrict__ vT16) {
  const int lane = threadIdx.x & 31;
  const int wave = threadIdx.x >> 5;
  const int wm = wave & 3, wn = wave >> 2;
  const int b  = blockIdx.z;
  const int n0 = blockIdx.x * 128 + wm * 32;   // key index tile
  const int o0 = blockIdx.y * 128 + wn * 64;   // output-channel tile
  const h16* A = x16 + (size_t)b * Nc * Cc;

  const h16* pA0 = frag_base(A, Cc, n0,      0, lane);
  const h16* pA1 = frag_base(A, Cc, n0 + 16, 0, lane);
  const h16* pB0 = frag_base(Wkv16, Cc, o0,      0, lane);
  const h16* pB1 = frag_base(Wkv16, Cc, o0 + 16, 0, lane);
  const h16* pB2 = frag_base(Wkv16, Cc, o0 + 32, 0, lane);
  const h16* pB3 = frag_base(Wkv16, Cc, o0 + 48, 0, lane);

  v8f zero = {};
  v8f acc[2][4] = {{zero, zero, zero, zero}, {zero, zero, zero, zero}};

  v16h a0 = load_frag_at(pA0);
  v16h a1 = load_frag_at(pA1);
  v16h b0 = load_frag_at(pB0);
  v16h b1 = load_frag_at(pB1);
  v16h b2 = load_frag_at(pB2);
  v16h b3 = load_frag_at(pB3);

#pragma unroll 4
  for (int k0 = 32; k0 < Cc; k0 += 32) {
    v16h na0 = load_frag_at(pA0 + k0);
    v16h na1 = load_frag_at(pA1 + k0);
    v16h nb0 = load_frag_at(pB0 + k0);
    v16h nb1 = load_frag_at(pB1 + k0);
    v16h nb2 = load_frag_at(pB2 + k0);
    v16h nb3 = load_frag_at(pB3 + k0);
    acc[0][0] = wmma16(a0, b0, acc[0][0]);
    acc[0][1] = wmma16(a0, b1, acc[0][1]);
    acc[0][2] = wmma16(a0, b2, acc[0][2]);
    acc[0][3] = wmma16(a0, b3, acc[0][3]);
    acc[1][0] = wmma16(a1, b0, acc[1][0]);
    acc[1][1] = wmma16(a1, b1, acc[1][1]);
    acc[1][2] = wmma16(a1, b2, acc[1][2]);
    acc[1][3] = wmma16(a1, b3, acc[1][3]);
    a0 = na0; a1 = na1; b0 = nb0; b1 = nb1; b2 = nb2; b3 = nb3;
  }
  acc[0][0] = wmma16(a0, b0, acc[0][0]);
  acc[0][1] = wmma16(a0, b1, acc[0][1]);
  acc[0][2] = wmma16(a0, b2, acc[0][2]);
  acc[0][3] = wmma16(a0, b3, acc[0][3]);
  acc[1][0] = wmma16(a1, b0, acc[1][0]);
  acc[1][1] = wmma16(a1, b1, acc[1][1]);
  acc[1][2] = wmma16(a1, b2, acc[1][2]);
  acc[1][3] = wmma16(a1, b3, acc[1][3]);

  const int hl = lane >> 4, cn = lane & 15;
  const int is_v = o0 >> 9;          // o0 in [512,1024) -> V
  const int hh   = (o0 & 511) >> 6;  // head; 64-wide tile == one head's D
#pragma unroll
  for (int mi = 0; mi < 2; ++mi) {
#pragma unroll
    for (int j = 0; j < 4; ++j) {
#pragma unroll
      for (int r = 0; r < 8; ++r) {
        const int n = n0 + 16 * mi + r + 8 * hl;
        const int d = 16 * j + cn;
        const h16 v = (h16)acc[mi][j][r];
        if (!is_v)
          k16[(((size_t)b * Hc + hh) * Nc + n) * Dc + d] = v;
        else
          vT16[(((size_t)b * Hc + hh) * Dc + d) * Nc + n] = v;
      }
    }
  }
}

// ---------------------------------------------------------------------------
// Flash attention.  4 waves/block share one (b,h): K/V chunks (32 keys) are
// staged once per block in double-buffered LDS via async global->LDS DMA
// (4x traffic reduction vs per-wave loads), overlapped with the softmax+WMMA
// of the previous chunk.  Q is pre-scaled by D^-1/2.
// ---------------------------------------------------------------------------
__global__ __launch_bounds__(128, 1) void flash_attn_wmma(
    const h16* __restrict__ qh16, const h16* __restrict__ k16,
    const h16* __restrict__ vT16, h16* __restrict__ ao16,
    const int* __restrict__ qlen) {
  __shared__ h16 kbuf[2][32 * 64];   // 32 keys x 64 d   (4KB each)
  __shared__ h16 vbuf[2][64 * 32];   // 64 d   x 32 keys (4KB each)
  __shared__ h16 pbuf[4 * 16 * 32];  // per-wave P tile

  const int tid  = threadIdx.x;
  const int lane = tid & 31;
  const int wave = tid >> 5;
  const int b = blockIdx.z, h = blockIdx.y;

  int off = 0;
  for (int i = 0; i < b; ++i) off += qlen[i];
  const int Lq = qlen[b];
  if (blockIdx.x * 64 >= Lq) return;   // block-uniform exit (Lq % 64 == 0),
                                       // keeps barriers balanced below
  const int row0 = blockIdx.x * 64 + wave * 16;

  const h16* Kb = k16  + ((size_t)b * Hc + h) * Nc * Dc;  // (N,D) rows=keys
  const h16* Vb = vT16 + ((size_t)b * Hc + h) * Dc * Nc;  // (D,N) rows=d

  v16h qa0 = load_frag(qh16, Cc, off + row0, h * Dc,      lane);
  v16h qa1 = load_frag(qh16, Cc, off + row0, h * Dc + 32, lane);

  float mrow[8], lrow[8];
#pragma unroll
  for (int r = 0; r < 8; ++r) { mrow[r] = -3.0e38f; lrow[r] = 0.0f; }
  v8f zero = {};
  v8f o[4] = {zero, zero, zero, zero};

  h16* my = pbuf + wave * 512;
  const int hl = lane >> 4, cn = lane & 15;

  // Stage one 32-key chunk (K: 4KB contiguous; V: 64 strided rows of 64B).
  // 4 x 16B copies per thread -> wait_asynccnt<4> leaves exactly the next
  // chunk's DMAs outstanding.
  auto stage = [&](int n0, int sel) {
    const h16* gk = Kb + (size_t)n0 * Dc;
    h16* lk = kbuf[sel];
    cp16_to_lds(gk + tid * 8,        lk + tid * 8);
    cp16_to_lds(gk + 1024 + tid * 8, lk + 1024 + tid * 8);
    h16* lv = vbuf[sel];
#pragma unroll
    for (int p = 0; p < 2; ++p) {
      const int c  = tid + p * 128;      // 16B chunk id, 256 total
      const int vr = c >> 2;
      const int vc = (c & 3) * 8;
      cp16_to_lds(Vb + (size_t)vr * Nc + n0 + vc, lv + vr * 32 + vc);
    }
  };

  constexpr int NCH = Nc / 32;
  stage(0, 0);
  for (int c = 0; c < NCH; ++c) {
    const int cur = c & 1;
    if (c + 1 < NCH) {
      stage((c + 1) * 32, cur ^ 1);
      wait_asynccnt<4>();   // chunk c complete; c+1 still in flight
    } else {
      wait_asynccnt<0>();
    }
    __syncthreads();

    const h16* kb = kbuf[cur];
    const h16* vb = vbuf[cur];

    // S = Q K^T : two 16x16 tiles, K-loop over D=64 (from LDS)
    v8f s0 = {}, s1 = {};
    s0 = wmma16(qa0, load_frag(kb, Dc, 0,  0,  lane), s0);
    s0 = wmma16(qa1, load_frag(kb, Dc, 0,  32, lane), s0);
    s1 = wmma16(qa0, load_frag(kb, Dc, 16, 0,  lane), s1);
    s1 = wmma16(qa1, load_frag(kb, Dc, 16, 32, lane), s1);

    // online softmax; accumulator row r <-> matrix row (r + 8*hl)
#pragma unroll
    for (int r = 0; r < 8; ++r) {
      const float mx = red16_max(fmaxf(s0[r], s1[r]));
      const float mn = fmaxf(mrow[r], mx);
      const float al = __expf(mrow[r] - mn);
      const float p0 = __expf(s0[r] - mn);
      const float p1 = __expf(s1[r] - mn);
      const float rs = red16_sum(p0 + p1);
      lrow[r] = lrow[r] * al + rs;
      mrow[r] = mn;
      o[0][r] *= al; o[1][r] *= al; o[2][r] *= al; o[3][r] *= al;
      const int prow = r + 8 * hl;
      my[prow * 32 + cn]      = (h16)p0;
      my[prow * 32 + 16 + cn] = (h16)p1;
    }

    // O += P (16x32 keys) * V-tile (d x keys), both from LDS
    v16h pa = load_frag(my, 32, 0, 0, lane);
    o[0] = wmma16(pa, load_frag(vb, 32, 0,  0, lane), o[0]);
    o[1] = wmma16(pa, load_frag(vb, 32, 16, 0, lane), o[1]);
    o[2] = wmma16(pa, load_frag(vb, 32, 32, 0, lane), o[2]);
    o[3] = wmma16(pa, load_frag(vb, 32, 48, 0, lane), o[3]);

    __syncthreads();   // all waves done reading buf before it is re-staged
  }

  float inv[8];
#pragma unroll
  for (int r = 0; r < 8; ++r) inv[r] = 1.0f / lrow[r];
#pragma unroll
  for (int j = 0; j < 4; ++j) {
#pragma unroll
    for (int r = 0; r < 8; ++r) {
      const size_t row = (size_t)(off + row0 + r + 8 * hl);
      ao16[row * Cc + h * Dc + 16 * j + cn] = (h16)(o[j][r] * inv[r]);
    }
  }
}

// ---------------------------------------------------------------------------
// Host orchestration
// ---------------------------------------------------------------------------
extern "C" void kernel_launch(void* const* d_in, const int* in_sizes, int n_in,
                              void* d_out, int out_size, void* d_ws,
                              size_t ws_size, hipStream_t stream) {
  const float* x     = (const float*)d_in[0];
  const float* q     = (const float*)d_in[1];
  const float* Wq    = (const float*)d_in[2];
  const float* Wkv   = (const float*)d_in[3];
  const float* Wproj = (const float*)d_in[4];
  const float* bproj = (const float*)d_in[5];
  const int*   qlen  = (const int*)d_in[6];
  const int TQ = in_sizes[1] / Cc;   // total query tokens (multiple of 128)

  char* ws = (char*)d_ws;
  size_t cur = 0;
  auto take = [&](size_t bytes) -> char* {
    char* p = ws + cur;
    cur = (cur + bytes + 255) & ~(size_t)255;
    return p;
  };
  h16* q16   = (h16*)take((size_t)TQ * Cc * 2);
  h16* x16   = (h16*)take((size_t)Bc * Nc * Cc * 2);
  h16* wq16  = (h16*)take((size_t)Cc * Cc * 2);
  h16* wkv16 = (h16*)take((size_t)2 * Cc * Cc * 2);
  h16* wp16  = (h16*)take((size_t)Cc * Cc * 2);
  h16* qh16  = (h16*)take((size_t)TQ * Cc * 2);
  h16* k16   = (h16*)take((size_t)Bc * Hc * Nc * Dc * 2);
  h16* vT16  = (h16*)take((size_t)Bc * Hc * Dc * Nc * 2);
  h16* ao16  = (h16*)take((size_t)TQ * Cc * 2);

  // 1) f32 -> f16 conversions
  f32_to_f16_kernel<<<1024, 256, 0, stream>>>(q, q16, TQ * Cc);
  f32_to_f16_kernel<<<2048, 256, 0, stream>>>(x, x16, Bc * Nc * Cc);
  f32_to_f16_kernel<<<256, 256, 0, stream>>>(Wq, wq16, Cc * Cc);
  f32_to_f16_kernel<<<512, 256, 0, stream>>>(Wkv, wkv16, 2 * Cc * Cc);
  f32_to_f16_kernel<<<256, 256, 0, stream>>>(Wproj, wp16, Cc * Cc);

  // 2) Q projection with softmax scale folded in: qh = (q @ Wq^T) * D^-1/2
  gemm_nt_wmma<<<dim3(TQ / 128, Cc / 128), 256, 0, stream>>>(
      q16, Cc, wq16, Cc, qh16, nullptr, Cc, nullptr, TQ, Cc, Cc, 0.125f);

  // 3) KV projection -> K (B,H,N,D) and V^T (B,H,D,N)
  kv_proj_wmma<<<dim3(Nc / 128, (2 * Cc) / 128, Bc), 256, 0, stream>>>(
      x16, wkv16, k16, vT16);

  // 4) Flash attention: 4 waves/block, LDS-shared async K/V chunks
  flash_attn_wmma<<<dim3(Nc / 64, Hc, Bc), 128, 0, stream>>>(
      qh16, k16, vT16, ao16, qlen);

  // 5) Output projection with bias (f32 result)
  gemm_nt_wmma<<<dim3(TQ / 128, Cc / 128), 256, 0, stream>>>(
      ao16, Cc, wp16, Cc, nullptr, (float*)d_out, Cc, bproj, TQ, Cc, Cc, 1.0f);
}
